// MultiheadLatentAttention_75694503625284
// MI455X (gfx1250) — matrix operand order repaired and checked by reading, not compiled, not run on hardware
//
#include <hip/hip_runtime.h>
#include <hip/hip_bf16.h>
#include <math.h>

// ---------------------------------------------------------------------------
// MLA with lightning-indexer top-k selection, absorbed-weight formulation.
// bf16 WMMA (v_wmma_f32_16x16x32_bf16) everywhere, f32 accumulation.
// CDNA5 data movement: global_load_async_to_lds_b128 for the latent gather,
// tensor_load_to_lds (TDM) for the per-token query tile, global_prefetch.
// ---------------------------------------------------------------------------

typedef __bf16 bf16_t;
typedef __attribute__((ext_vector_type(16))) __bf16        v16bf;
typedef __attribute__((ext_vector_type(8)))  float         v8f;
typedef __attribute__((ext_vector_type(4)))  unsigned int  v4u;
typedef __attribute__((ext_vector_type(8)))  int           i32x8;

union Frag { v16bf v; v4u q[2]; bf16_t h[16]; };

#define S_LEN  1024
#define DMODEL 1024
#define DCKV   256
#define DCQ    512
#define NH     8
#define DH     64
#define DR     32
#define KSEL   128
#define LWIN   64
#define DCAT   (DCKV + DR)   // 288 = concat(latent 256, rope 32)

static __device__ __forceinline__ v8f wmma_bf16(v16bf a, v16bf b, v8f c) {
  // D = A(16x32) * B(32x16) + C(16x16 f32)
  return __builtin_amdgcn_wmma_f32_16x16x32_bf16(false, a, false, b, (short)0, c,
                                                 false, false);
}

static __device__ __forceinline__ float softplusf_(float x) {
  return (x > 20.f) ? x : log1pf(expf(x));
}
static __device__ __forceinline__ float sigmoidf_(float x) {
  return 1.f / (1.f + expf(-x));
}

// 16-byte global->LDS async copy (ASYNCcnt-tracked DMA, no VGPR staging)
static __device__ __forceinline__ void async_copy16(unsigned int lds_off,
                                                    const void* gptr) {
  asm volatile("global_load_async_to_lds_b128 %0, %1, off"
               :: "v"(lds_off), "v"((unsigned long long)(size_t)gptr)
               : "memory");
}
static __device__ __forceinline__ void wait_async() {
  asm volatile("s_wait_asynccnt 0x0" ::: "memory");
}

// ---------------------------------------------------------------------------
// elementwise casts
// ---------------------------------------------------------------------------
__global__ void k_cast(const float* __restrict__ s, bf16_t* __restrict__ d, int n) {
  int i = blockIdx.x * blockDim.x + threadIdx.x;
  if (i < n) d[i] = (bf16_t)s[i];
}

// src (R x C) f32 row-major  ->  dst (C x R) bf16 row-major (i.e. srcT)
__global__ void k_tcast(const float* __restrict__ s, bf16_t* __restrict__ d, int R, int C) {
  int i = blockIdx.x * blockDim.x + threadIdx.x;
  if (i < R * C) {
    int r = i / C, c = i % C;
    d[(size_t)c * R + r] = (bf16_t)s[i];
  }
}

// weffT[h][c][q] = sum_d w_uk[c, h*64+d] * w_uq[q, h*64+d]   (= (w_uq_h @ w_uk_h^T)^T)
__global__ void k_weffT(const float* __restrict__ w_uk, const float* __restrict__ w_uq,
                        bf16_t* __restrict__ weffT) {
  int i = blockIdx.x * blockDim.x + threadIdx.x;
  if (i >= NH * DCKV * DCQ) return;
  int h = i / (DCKV * DCQ);
  int rem = i % (DCKV * DCQ);
  int c = rem / DCQ;
  int q = rem % DCQ;
  float acc = 0.f;
#pragma unroll 8
  for (int d = 0; d < DH; ++d)
    acc += w_uk[(size_t)c * (NH * DH) + h * DH + d] *
           w_uq[(size_t)q * (NH * DH) + h * DH + d];
  weffT[i] = (bf16_t)acc;
}

// ---------------------------------------------------------------------------
// generic WMMA GEMM: C(f32, M x N) = A(bf16, M x K) * Bt(bf16, N x K)^T
// one wave computes a 16 x (16*NT) strip: one A fragment feeds NT WMMAs per
// k-step (A-reuse, independent accumulators for XDL latency hiding).
// grid = (N/(16*NT), M/16), block = 32.
// Fragment k order: contiguous, lane half h holds k = 16h..16h+15 (A and B alike).
// ---------------------------------------------------------------------------
template <int NT>
__global__ void k_gemm_t(const bf16_t* __restrict__ A, int lda,
                         const bf16_t* __restrict__ Bt, int ldb,
                         float* __restrict__ C, int ldc, int K) {
  const int lane  = threadIdx.x;
  const int tileN = blockIdx.x * 16 * NT, tileM = blockIdx.y * 16;
  const int mn  = lane & 15;
  const int kbA = (lane >> 4) * 16;
  const int kbB = (lane >> 4) * 16;
  const bf16_t* arow = A + (size_t)(tileM + mn) * lda;
  const bf16_t* brow[NT];
#pragma unroll
  for (int j = 0; j < NT; ++j)
    brow[j] = Bt + (size_t)(tileN + j * 16 + mn) * ldb;
  v8f acc[NT];
#pragma unroll
  for (int j = 0; j < NT; ++j)
    acc[j] = (v8f){0.f, 0.f, 0.f, 0.f, 0.f, 0.f, 0.f, 0.f};
  for (int k0 = 0; k0 < K; k0 += 32) {
    __builtin_prefetch(arow + k0 + 128, 0, 0);   // global_prefetch_b8, 4 tiles ahead
    Frag a;
    a.q[0] = *(const v4u*)(arow + k0 + kbA);
    a.q[1] = *(const v4u*)(arow + k0 + kbA + 8);
#pragma unroll
    for (int j = 0; j < NT; ++j) {
      Frag b;
      b.q[0] = *(const v4u*)(brow[j] + k0 + kbB);
      b.q[1] = *(const v4u*)(brow[j] + k0 + kbB + 8);
      acc[j] = wmma_bf16(a.v, b.v, acc[j]);
    }
  }
  const int row0 = tileM + ((lane >> 4) ? 8 : 0);
#pragma unroll
  for (int j = 0; j < NT; ++j)
#pragma unroll
    for (int r = 0; r < 8; ++r)
      C[(size_t)(row0 + r) * ldc + tileN + j * 16 + mn] = acc[j][r];
}

// ---------------------------------------------------------------------------
// Lightning indexer: I[t,s] = sum_h w_idx[h] * relu(dot(qI[t,h,:], kI[s,:]))
// A = qIb (S x 512), Bt = kIb (S x 64) directly (kI rows ARE B^T rows).
// ---------------------------------------------------------------------------
__global__ void k_indexer(const bf16_t* __restrict__ qIb, const bf16_t* __restrict__ kIb,
                          const float* __restrict__ w_idx, float* __restrict__ I) {
  const int lane  = threadIdx.x;
  const int tileN = blockIdx.x * 16, tileM = blockIdx.y * 16;  // N = key pos, M = query pos
  const int mn  = lane & 15;
  const int kbA = (lane >> 4) * 16;
  const int kbB = (lane >> 4) * 16;
  v8f racc = {0.f, 0.f, 0.f, 0.f, 0.f, 0.f, 0.f, 0.f};
  for (int h = 0; h < NH; ++h) {
    const bf16_t* arow = qIb + (size_t)(tileM + mn) * (NH * DH) + h * DH;
    const bf16_t* brow = kIb + (size_t)(tileN + mn) * DH;
    v8f acc = {0.f, 0.f, 0.f, 0.f, 0.f, 0.f, 0.f, 0.f};
#pragma unroll
    for (int k0 = 0; k0 < DH; k0 += 32) {
      Frag a, b;
      a.q[0] = *(const v4u*)(arow + k0 + kbA);
      a.q[1] = *(const v4u*)(arow + k0 + kbA + 8);
      b.q[0] = *(const v4u*)(brow + k0 + kbB);
      b.q[1] = *(const v4u*)(brow + k0 + kbB + 8);
      acc = wmma_bf16(a.v, b.v, acc);
    }
    float wh = w_idx[h];
#pragma unroll
    for (int r = 0; r < 8; ++r) racc[r] += wh * fmaxf(acc[r], 0.f);
  }
  const int row0 = tileM + ((lane >> 4) ? 8 : 0);
#pragma unroll
  for (int r = 0; r < 8; ++r)
    I[(size_t)(row0 + r) * S_LEN + tileN + mn] = racc[r];
}

// ---------------------------------------------------------------------------
// Top-k per row: local window -> +big, causal future -> -big, 128 iterative
// argmax picks (ascending-index tie-break, mimicking jax.lax.top_k).
// ---------------------------------------------------------------------------
__global__ void k_topk(const float* __restrict__ I, int* __restrict__ sel) {
  __shared__ float sc[S_LEN];
  __shared__ float rv[128];
  __shared__ int   ri[128];
  const int t = blockIdx.x, tid = threadIdx.x;
  for (int s = tid; s < S_LEN; s += 128) {
    float v = I[(size_t)t * S_LEN + s];
    if (s > t)                  v = -1e37f;  // causal (still selectable when row short)
    else if (s >= t - (LWIN-1)) v =  1e37f;  // forced local window
    sc[s] = v;
  }
  __syncthreads();
  for (int it = 0; it < KSEL; ++it) {
    float bv = -2e38f; int bi = 0;
    for (int s = tid; s < S_LEN; s += 128) {
      float v = sc[s];
      if (v > bv) { bv = v; bi = s; }
    }
    rv[tid] = bv; ri[tid] = bi;
    __syncthreads();
    for (int off = 64; off > 0; off >>= 1) {
      if (tid < off) {
        if (rv[tid + off] > rv[tid] ||
            (rv[tid + off] == rv[tid] && ri[tid + off] < ri[tid])) {
          rv[tid] = rv[tid + off]; ri[tid] = ri[tid + off];
        }
      }
      __syncthreads();
    }
    if (tid == 0) { sel[(size_t)t * KSEL + it] = ri[0]; sc[ri[0]] = -3e38f; }
    __syncthreads();
  }
}

// ---------------------------------------------------------------------------
// PoPE for query-rope (position = query index t)
// ---------------------------------------------------------------------------
__global__ void k_qrpope(const float* __restrict__ qr_raw, const float* __restrict__ raw_delta,
                         float* __restrict__ qrp) {
  const int t = blockIdx.x, tid = threadIdx.x;     // 256 threads = 8 heads x 32
  const int h = tid >> 5, i = tid & 31, j = i & 15;
  float v1 = qr_raw[(size_t)t * (NH * DR) + h * DR + j];
  float v2 = qr_raw[(size_t)t * (NH * DR) + h * DR + 16 + j];
  float mu1 = softplusf_(v1), mu2 = softplusf_(v2);
  float th  = powf(10000.f, -(float)j / 16.f);
  float dl  = -6.2831853071795864f * sigmoidf_(raw_delta[j]);
  float ang = (float)t * th + dl;
  float c = cosf(ang), s = sinf(ang);
  qrp[(size_t)(t * NH + h) * DR + i] = (i < 16) ? (mu1 * c - mu2 * s)
                                                : (mu1 * s + mu2 * c);
}

// ---------------------------------------------------------------------------
// Fused attention: per query token t (block = 128 threads, wave0 computes).
// Async-DMA gathers 128 selected latents + slot-PoPE rope keys into LDS in
// two 64-slot halves; TDM loads the q_eff tile; WMMA scores (K=288), softmax,
// WMMA latent accumulation (K=128), writes o_lat (h, t, 256) bf16.
// ---------------------------------------------------------------------------
static __device__ __forceinline__ void gather_half(
    bf16_t (*Kc)[DCAT], const int* __restrict__ sel, const bf16_t* __restrict__ ckvb,
    const float* __restrict__ krp, const float* __restrict__ raw_delta,
    int t, int half, int tid) {
  const int jl = tid >> 1, p = tid & 1;          // 2 threads per selected slot
  const int token = sel[(size_t)t * KSEL + half * 64 + jl];
  // 128 bf16 of latent via ASYNCcnt-tracked LDS DMA (16 x b128)
  const unsigned int lbase = (unsigned int)(size_t)(&Kc[jl][p * 128]);
  const char* gsrc = (const char*)(ckvb + (size_t)token * DCKV + p * 128);
#pragma unroll
  for (int q = 0; q < 16; ++q)
    async_copy16(lbase + q * 16, gsrc + q * 16);
  // overlap: compute slot-position PoPE rope keys on the VALU while DMA lands
  const int pos = half * 64 + jl;                // PoPE position = slot index
#pragma unroll
  for (int ii = 0; ii < 8; ++ii) {
    int i = p * 8 + ii;
    float x1 = krp[(size_t)token * DR + i];
    float x2 = krp[(size_t)token * DR + 16 + i];
    float mu1 = softplusf_(x1), mu2 = softplusf_(x2);
    float th  = powf(10000.f, -(float)i / 16.f);
    float dl  = -6.2831853071795864f * sigmoidf_(raw_delta[i]);
    float ang = (float)pos * th + dl;
    float cs = cosf(ang), sn = sinf(ang);
    Kc[jl][DCKV + i]      = (bf16_t)(mu1 * cs - mu2 * sn);
    Kc[jl][DCKV + 16 + i] = (bf16_t)(mu1 * sn + mu2 * cs);
  }
  wait_async();
}

__global__ void k_attn(const int* __restrict__ sel,
                       const bf16_t* __restrict__ ckvb,
                       const float* __restrict__ krp,
                       const float* __restrict__ qeff,
                       const float* __restrict__ qrp,
                       const float* __restrict__ raw_delta,
                       bf16_t* __restrict__ olatb) {
  __shared__ bf16_t Kc[64][DCAT];       // 36864 B : gathered keys (one half)
  __shared__ bf16_t qc[16][DCAT];       //  9216 B : q_eff|q_rope, heads padded to 16
  __shared__ float  attnS[NH][KSEL];    //  4096 B : raw scores / exp
  __shared__ bf16_t attnB[16][KSEL];    //  4096 B : normalized probs (rows 8-15 = 0)
  __shared__ float  olat[NH][DCKV];     //  8192 B : q_eff TDM staging, then o_lat acc
  const int t = blockIdx.x, tid = threadIdx.x, lane = tid & 31;

  // ---- TDM: 2D tensor tile (8 rows x 256 f32, row stride S*256) -> LDS ----
  if (tid < 32) {
    const unsigned long long ga =
        (unsigned long long)(size_t)(qeff + (size_t)t * DCKV);
    const unsigned int lof = (unsigned int)(size_t)(&olat[0][0]);
    v4u g0;
    g0.x = 1u;                                        // count=1, user mode
    g0.y = lof;                                       // lds_addr
    g0.z = (unsigned int)(ga & 0xffffffffu);          // global_addr[31:0]
    g0.w = (unsigned int)((ga >> 32) & 0x1ffffffu) |  // global_addr[56:32]
           (2u << 30);                                // type=2 (image)
    const unsigned long long st0 = (unsigned long long)S_LEN * DCKV; // dim0 stride
    i32x8 g1;
    g1[0] = (int)(2u << 16);                          // data_size=4B, no mask/pad
    g1[1] = (int)((unsigned)(DCKV & 0xffff) << 16);   // tensor_dim0[15:0]
    g1[2] = (int)((DCKV >> 16) | ((unsigned)NH << 16)); // dim0 hi | tensor_dim1 lo
    g1[3] = (int)((unsigned)DCKV << 16);              // tensor_dim1 hi=0 | tile_dim0
    g1[4] = (int)NH;                                  // tile_dim1=8, tile_dim2=0
    g1[5] = (int)(unsigned int)(st0 & 0xffffffffu);   // tensor_dim0_stride[31:0]
    g1[6] = (int)(unsigned int)(st0 >> 32);           // stride hi | dim1_stride lo=0
    g1[7] = 0;
    asm volatile("tensor_load_to_lds %0, %1" :: "s"(g0), "s"(g1) : "memory");
    __builtin_amdgcn_s_wait_tensorcnt(0);
  }
  __syncthreads();

  // build q_cat from staged q_eff + q_rope; zero attn pad rows
  for (int i = tid; i < 16 * DCAT; i += 128) {
    int m = i / DCAT, c = i % DCAT;
    float v = 0.f;
    if (m < NH)
      v = (c < DCKV) ? olat[m][c]
                     : qrp[((size_t)t * NH + m) * DR + (c - DCKV)];
    qc[m][c] = (bf16_t)v;
  }
  for (int i = tid; i < 16 * KSEL; i += 128) attnB[i / KSEL][i % KSEL] = (bf16_t)0.f;
  __syncthreads();
  for (int i = tid; i < NH * DCKV; i += 128) olat[i / DCKV][i % DCKV] = 0.f;

  const int mn  = lane & 15;
  const int kbA = (lane >> 4) * 16;
  const int kbB = (lane >> 4) * 16;
  const float scale = 0.1020620726159658f;   // (DH+DR)^-0.5 = 1/sqrt(96)

  // ---------------- pass 1: scores ----------------
  for (int half = 0; half < 2; ++half) {
    __syncthreads();
    gather_half(Kc, sel, ckvb, krp, raw_delta, t, half, tid);
    __syncthreads();
    if (tid < 32) {
      for (int nt = 0; nt < 4; ++nt) {           // 4 x 16 slots per half
        v8f acc = {0.f, 0.f, 0.f, 0.f, 0.f, 0.f, 0.f, 0.f};
#pragma unroll
        for (int ks = 0; ks < 9; ++ks) {         // K = 288
          Frag a, b;
          a.q[0] = *(const v4u*)(&qc[mn][ks * 32 + kbA]);
          a.q[1] = *(const v4u*)(&qc[mn][ks * 32 + kbA + 8]);
          b.q[0] = *(const v4u*)(&Kc[nt * 16 + mn][ks * 32 + kbB]);
          b.q[1] = *(const v4u*)(&Kc[nt * 16 + mn][ks * 32 + kbB + 8]);
          acc = wmma_bf16(a.v, b.v, acc);
        }
        if (lane < 16) {                         // lanes 0-15 hold heads 0-7
#pragma unroll
          for (int r = 0; r < NH; ++r)
            attnS[r][half * 64 + nt * 16 + lane] = acc[r] * scale;
        }
      }
    }
  }
  __syncthreads();

  // ---------------- softmax over 128 slots per head (wave0) ----------------
  if (tid < 32) {
    for (int hh = 0; hh < NH; ++hh) {
      float mx = -1e30f;
      for (int j = lane; j < KSEL; j += 32) mx = fmaxf(mx, attnS[hh][j]);
#pragma unroll
      for (int off = 16; off; off >>= 1) mx = fmaxf(mx, __shfl_xor(mx, off, 32));
      float sum = 0.f;
      for (int j = lane; j < KSEL; j += 32) {
        float e = expf(attnS[hh][j] - mx);
        attnS[hh][j] = e; sum += e;
      }
#pragma unroll
      for (int off = 16; off; off >>= 1) sum += __shfl_xor(sum, off, 32);
      float inv = 1.f / sum;
      for (int j = lane; j < KSEL; j += 32)
        attnB[hh][j] = (bf16_t)(attnS[hh][j] * inv);
    }
  }

  // ---------------- pass 2: o_lat = attn @ gathered latents ----------------
  for (int half = 0; half < 2; ++half) {
    __syncthreads();
    gather_half(Kc, sel, ckvb, krp, raw_delta, t, half, tid);
    __syncthreads();
    if (tid < 32) {
      for (int ct = 0; ct < 16; ++ct) {          // 16 x 16-channel tiles of 256
        v8f acc = {0.f, 0.f, 0.f, 0.f, 0.f, 0.f, 0.f, 0.f};
#pragma unroll
        for (int ks = 0; ks < 2; ++ks) {         // 64 slots per half
          Frag a, b;
          int k0 = half * 64 + ks * 32;
          a.q[0] = *(const v4u*)(&attnB[mn][k0 + kbA]);
          a.q[1] = *(const v4u*)(&attnB[mn][k0 + kbA + 8]);
#pragma unroll
          for (int e = 0; e < 16; ++e)           // strided column read of V
            b.h[e] = Kc[ks * 32 + kbB + e][ct * 16 + mn];
          acc = wmma_bf16(a.v, b.v, acc);
        }
        if (lane < 16) {
#pragma unroll
          for (int r = 0; r < NH; ++r) olat[r][ct * 16 + lane] += acc[r];
        }
      }
    }
  }
  __syncthreads();
  for (int i = tid; i < NH * DCKV; i += 128) {
    int hh = i / DCKV, c = i % DCKV;
    olatb[((size_t)hh * S_LEN + t) * DCKV + c] = (bf16_t)olat[hh][c];
  }
}

// ---------------------------------------------------------------------------
// host launcher
// ---------------------------------------------------------------------------
extern "C" void kernel_launch(void* const* d_in, const int* in_sizes, int n_in,
                              void* d_out, int out_size, void* d_ws, size_t ws_size,
                              hipStream_t stream) {
  (void)in_sizes; (void)n_in; (void)out_size; (void)ws_size;
  const float* x         = (const float*)d_in[0];
  const float* w_q_idx   = (const float*)d_in[1];
  const float* w_k_idx   = (const float*)d_in[2];
  const float* w_idx     = (const float*)d_in[3];
  const float* raw_delta = (const float*)d_in[4];
  const float* w_dkv     = (const float*)d_in[5];
  const float* w_uk      = (const float*)d_in[6];
  const float* w_uv      = (const float*)d_in[7];
  const float* w_dq      = (const float*)d_in[8];
  const float* w_uq      = (const float*)d_in[9];
  const float* w_qr      = (const float*)d_in[10];
  const float* w_kr      = (const float*)d_in[11];
  const float* w_out     = (const float*)d_in[12];
  float* out = (float*)d_out;

  char* wsb = (char*)d_ws;
  size_t off = 0;
  auto alloc = [&](size_t bytes) -> void* {
    void* p = wsb + off;
    off = (off + bytes + 255) & ~(size_t)255;
    return p;
  };

  // bf16 scratch
  bf16_t* xb     = (bf16_t*)alloc(2ull * S_LEN * DMODEL);
  bf16_t* wqT    = (bf16_t*)alloc(2ull * (NH*DH) * DMODEL);
  bf16_t* wkT    = (bf16_t*)alloc(2ull * DH * DMODEL);
  bf16_t* wdkvT  = (bf16_t*)alloc(2ull * DCKV * DMODEL);
  bf16_t* wdqT   = (bf16_t*)alloc(2ull * DCQ * DMODEL);
  bf16_t* wkrT   = (bf16_t*)alloc(2ull * DR * DMODEL);
  bf16_t* wqrT   = (bf16_t*)alloc(2ull * (NH*DR) * DCQ);
  bf16_t* wuvT   = (bf16_t*)alloc(2ull * (NH*DH) * DCKV);
  bf16_t* woutT  = (bf16_t*)alloc(2ull * DMODEL * (NH*DH));
  bf16_t* weffT  = (bf16_t*)alloc(2ull * NH * DCKV * DCQ);
  bf16_t* qIb    = (bf16_t*)alloc(2ull * S_LEN * (NH*DH));
  bf16_t* kIb    = (bf16_t*)alloc(2ull * S_LEN * DH);
  bf16_t* ckvb   = (bf16_t*)alloc(2ull * S_LEN * DCKV);
  bf16_t* cqb    = (bf16_t*)alloc(2ull * S_LEN * DCQ);
  bf16_t* ob     = (bf16_t*)alloc(2ull * S_LEN * (NH*DH));
  bf16_t* olatb  = (bf16_t*)alloc(2ull * NH * S_LEN * DCKV);
  // f32 scratch
  float* qIf   = (float*)alloc(4ull * S_LEN * (NH*DH));
  float* kIf   = (float*)alloc(4ull * S_LEN * DH);
  float* ckvf  = (float*)alloc(4ull * S_LEN * DCKV);
  float* cqf   = (float*)alloc(4ull * S_LEN * DCQ);
  float* krpf  = (float*)alloc(4ull * S_LEN * DR);
  float* qrf   = (float*)alloc(4ull * S_LEN * (NH*DR));
  float* qrpf  = (float*)alloc(4ull * S_LEN * (NH*DR));
  float* qefff = (float*)alloc(4ull * NH * S_LEN * DCKV);
  float* If    = (float*)alloc(4ull * S_LEN * S_LEN);
  float* of    = (float*)alloc(4ull * S_LEN * (NH*DH));
  int*   sel   = (int*)alloc(4ull * S_LEN * KSEL);

  auto blk = [](int n) { return dim3((unsigned)((n + 255) / 256)); };
  auto gemm = [&](const bf16_t* A, int lda, const bf16_t* Bt, int ldb,
                  float* C, int ldc, int M, int N, int K) {
    if (N % 64 == 0) {
      dim3 g((unsigned)(N / 64), (unsigned)(M / 16));
      k_gemm_t<4><<<g, dim3(32), 0, stream>>>(A, lda, Bt, ldb, C, ldc, K);
    } else {
      dim3 g((unsigned)(N / 32), (unsigned)(M / 16));
      k_gemm_t<2><<<g, dim3(32), 0, stream>>>(A, lda, Bt, ldb, C, ldc, K);
    }
  };

  // casts + weight transposes
  k_cast<<<blk(S_LEN*DMODEL), 256, 0, stream>>>(x, xb, S_LEN*DMODEL);
  k_tcast<<<blk(DMODEL*(NH*DH)), 256, 0, stream>>>(w_q_idx, wqT,  DMODEL, NH*DH);
  k_tcast<<<blk(DMODEL*DH),      256, 0, stream>>>(w_k_idx, wkT,  DMODEL, DH);
  k_tcast<<<blk(DMODEL*DCKV),    256, 0, stream>>>(w_dkv,  wdkvT, DMODEL, DCKV);
  k_tcast<<<blk(DMODEL*DCQ),     256, 0, stream>>>(w_dq,   wdqT,  DMODEL, DCQ);
  k_tcast<<<blk(DMODEL*DR),      256, 0, stream>>>(w_kr,   wkrT,  DMODEL, DR);
  k_tcast<<<blk(DCQ*(NH*DR)),    256, 0, stream>>>(w_qr,   wqrT,  DCQ,    NH*DR);
  k_tcast<<<blk(DCKV*(NH*DH)),   256, 0, stream>>>(w_uv,   wuvT,  DCKV,   NH*DH);
  k_tcast<<<blk((NH*DH)*DMODEL), 256, 0, stream>>>(w_out,  woutT, NH*DH,  DMODEL);
  k_weffT<<<blk(NH*DCKV*DCQ),    256, 0, stream>>>(w_uk, w_uq, weffT);

  // projections from x
  gemm(xb, DMODEL, wqT,   DMODEL, qIf,  NH*DH, S_LEN, NH*DH, DMODEL);
  k_cast<<<blk(S_LEN*(NH*DH)), 256, 0, stream>>>(qIf, qIb, S_LEN*(NH*DH));
  gemm(xb, DMODEL, wkT,   DMODEL, kIf,  DH,    S_LEN, DH,    DMODEL);
  k_cast<<<blk(S_LEN*DH), 256, 0, stream>>>(kIf, kIb, S_LEN*DH);
  gemm(xb, DMODEL, wdkvT, DMODEL, ckvf, DCKV,  S_LEN, DCKV,  DMODEL);
  k_cast<<<blk(S_LEN*DCKV), 256, 0, stream>>>(ckvf, ckvb, S_LEN*DCKV);
  gemm(xb, DMODEL, wdqT,  DMODEL, cqf,  DCQ,   S_LEN, DCQ,   DMODEL);
  k_cast<<<blk(S_LEN*DCQ), 256, 0, stream>>>(cqf, cqb, S_LEN*DCQ);
  gemm(xb, DMODEL, wkrT,  DMODEL, krpf, DR,    S_LEN, DR,    DMODEL);

  // query-side: rope + absorbed effective queries
  gemm(cqb, DCQ, wqrT, DCQ, qrf, NH*DR, S_LEN, NH*DR, DCQ);
  k_qrpope<<<dim3(S_LEN), dim3(256), 0, stream>>>(qrf, raw_delta, qrpf);
  for (int h = 0; h < NH; ++h)
    gemm(cqb, DCQ, weffT + (size_t)h*DCKV*DCQ, DCQ,
         qefff + (size_t)h*S_LEN*DCKV, DCKV, S_LEN, DCKV, DCQ);

  // indexer + top-k selection
  k_indexer<<<dim3(S_LEN/16, S_LEN/16), dim3(32), 0, stream>>>(qIb, kIb, w_idx, If);
  k_topk<<<dim3(S_LEN), dim3(128), 0, stream>>>(If, sel);

  // fused gather / scores / softmax / latent output
  k_attn<<<dim3(S_LEN), dim3(128), 0, stream>>>(sel, ckvb, krpf, qefff, qrpf,
                                                raw_delta, olatb);

  // per-head up-projection (w_uv) then output projection (w_out)
  for (int h = 0; h < NH; ++h)
    gemm(olatb + (size_t)h*S_LEN*DCKV, DCKV,
         wuvT + (size_t)h*DH*DCKV, DCKV,
         of + h*DH, NH*DH, S_LEN, DH, DCKV);
  k_cast<<<blk(S_LEN*(NH*DH)), 256, 0, stream>>>(of, ob, S_LEN*(NH*DH));
  gemm(ob, NH*DH, woutT, NH*DH, out, DMODEL, S_LEN, DMODEL, NH*DH);
}
